// SimpleGRU_11974368821377
// MI455X (gfx1250) — matrix operand order repaired and checked by reading, not compiled, not run on hardware
//
#include <hip/hip_runtime.h>

// ---------------------------------------------------------------------------
// Fused GRU for MI455X (gfx1250, wave32, WMMA), round 2.
//   B=4096, T=256, IN=64, H=20 (gates 3H=60 padded to 64; h-GEMM K 20->32)
// 256 WGs x 128 threads (4 wave32s, one 16x16 gate tile per wave).
// x streamed from HBM exactly once (256 MB ~ 11us floor at 23.3 TB/s),
// double-buffered through LDS so the stream overlaps WMMA + gate math.
// h persists in registers (fp32 master) + LDS (bf16 WMMA operand).
// Gate nonlinearities use hardware v_tanh_f32 when available.
// ---------------------------------------------------------------------------

typedef __attribute__((ext_vector_type(16))) __bf16 v16bf;
typedef __attribute__((ext_vector_type(4)))  __bf16 bf16x4;
typedef __attribute__((ext_vector_type(8)))  float  v8f;

#define B_TOT   4096
#define T_LEN   256
#define IN_D    64
#define HID     20
#define G3      60
#define MT      16      // batch rows per workgroup
#define NTHR    128     // 4 wave32s

#define XPAD    72      // bf16 row stride (144 B): bank rotation + 16B runs aligned
#define HPAD    40      // bf16 row stride (80 B)
#define GPAD    68      // f32 row stride

__device__ __forceinline__ float tanh_fast(float v) {
#if __has_builtin(__builtin_amdgcn_tanhf)
    return __builtin_amdgcn_tanhf(v);          // v_tanh_f32 (TRANS)
#else
    float e = __expf(2.f * v);                 // v_exp_f32
    return 1.f - 2.f * __builtin_amdgcn_rcpf(e + 1.f);  // v_rcp_f32
#endif
}
__device__ __forceinline__ float sigmoid_fast(float v) {
    return 0.5f + 0.5f * tanh_fast(0.5f * v);
}

__global__ __launch_bounds__(NTHR)
void gru_fused_wmma(const float* __restrict__ x,
                    const float* __restrict__ W_ih,   // [60,64]
                    const float* __restrict__ W_hh,   // [60,20]
                    const float* __restrict__ b_ih,   // [60]
                    const float* __restrict__ b_hh,   // [60]
                    const float* __restrict__ W_head, // [1,20]
                    const float* __restrict__ b_head, // [1]
                    float* __restrict__ out)          // [4096]
{
    __shared__ __attribute__((aligned(16))) __bf16 xb[2][MT][XPAD]; // double-buffered x_t
    __shared__ __attribute__((aligned(16))) __bf16 hb[MT][HPAD];    // h (bf16, cols 20..31 = 0)
    __shared__ __attribute__((aligned(16))) float  gx_s[MT][GPAD];
    __shared__ __attribute__((aligned(16))) float  gh_s[MT][GPAD];
    __shared__ float hf[MT][HID + 1];                               // final h for head proj

    const int tid   = threadIdx.x;
    const int lane  = tid & 31;
    const int wave  = tid >> 5;       // 0..3 -> gate column tile
    const int col   = lane & 15;
    const int khalf = lane >> 4;
    const int b0    = blockIdx.x * MT;

    // -------- loop-invariant weight B-fragments (gathered once) ------------
    // B layout (16-bit, Kx16): lane = column N, element e -> K = 8*khalf + e (e<8), 16 + 8*khalf + (e-8) (e>=8).
    v16bf bw_ih0, bw_ih1, bw_hh;
    {
        const int n = wave * 16 + col;
        const bool nv = (n < G3);
        #pragma unroll
        for (int e = 0; e < 16; ++e) {
            const int k = (e < 8) ? (khalf * 8 + e) : (16 + khalf * 8 + (e - 8));
            bw_ih0[e] = (__bf16)(nv ? W_ih[n * IN_D + k]      : 0.f);
            bw_ih1[e] = (__bf16)(nv ? W_ih[n * IN_D + 32 + k] : 0.f);
            bw_hh[e]  = (__bf16)((nv && k < HID) ? W_hh[n * HID + k] : 0.f);
        }
    }

    // -------- per-thread gate items: i = tid, tid+128, tid+256 (<320) ------
    const int  r0 = tid / HID,          c0 = tid % HID;
    const int  r1 = (tid + 128) / HID,  c1 = (tid + 128) % HID;
    const int  r2 = (tid + 256) / HID,  c2 = (tid + 256) % HID;
    const bool has2 = (tid < 64);
    // pre-summed biases (r,z gates) + separate n-gate biases
    const float br0 = b_ih[c0] + b_hh[c0],           br1 = b_ih[c1] + b_hh[c1];
    const float bz0 = b_ih[c0+HID] + b_hh[c0+HID],   bz1 = b_ih[c1+HID] + b_hh[c1+HID];
    const float bni0 = b_ih[c0+2*HID], bnh0 = b_hh[c0+2*HID];
    const float bni1 = b_ih[c1+2*HID], bnh1 = b_hh[c1+2*HID];
    const float br2  = has2 ? b_ih[c2] + b_hh[c2] : 0.f;
    const float bz2  = has2 ? b_ih[c2+HID] + b_hh[c2+HID] : 0.f;
    const float bni2 = has2 ? b_ih[c2+2*HID] : 0.f;
    const float bnh2 = has2 ? b_hh[c2+2*HID] : 0.f;
    float h0 = 0.f, h1 = 0.f, h2 = 0.f;   // fp32 h master, in registers

    // -------- x staging pointers: thread owns 2 float4 chunks per step -----
    const int ra = tid >> 4, rb = ra + 8;
    const int cc = (tid & 15) * 4;
    const float* pa = x + ((size_t)(b0 + ra) * T_LEN) * IN_D + cc;
    const float* pb = x + ((size_t)(b0 + rb) * T_LEN) * IN_D + cc;

    // zero hb (incl. K padding), stage t=0 into buffer 0
    for (int i = tid; i < MT * HPAD; i += NTHR) hb[i / HPAD][i % HPAD] = (__bf16)0.f;
    {
        float4 va = *(const float4*)pa;
        float4 vb = *(const float4*)pb;
        *(bf16x4*)&xb[0][ra][cc] = (bf16x4){(__bf16)va.x,(__bf16)va.y,(__bf16)va.z,(__bf16)va.w};
        *(bf16x4*)&xb[0][rb][cc] = (bf16x4){(__bf16)vb.x,(__bf16)vb.y,(__bf16)vb.z,(__bf16)vb.w};
    }
    __syncthreads();

    for (int t = 0; t < T_LEN; ++t) {
        const int cur = t & 1, nxt = cur ^ 1;
        const bool more = (t + 1 < T_LEN);   // uniform branch (EXEC stays all-1s)

        // ---- issue next step's global loads early (land at end barrier) ---
        float4 va, vb;
        if (more) {
            const float* na = pa + (size_t)(t + 1) * IN_D;
            const float* nb = pb + (size_t)(t + 1) * IN_D;
            va = *(const float4*)na;
            vb = *(const float4*)nb;
            __builtin_prefetch(na + IN_D, 0, 3);   // t+2
            __builtin_prefetch(nb + IN_D, 0, 3);
        }

        // ---- A fragments (16-bit 16x32 layout; two aligned 16B runs) ------
        v16bf a0, a1, ah;
        {
            const __bf16* xr = &xb[cur][col][0];
            const __bf16* hr = &hb[col][0];
            #pragma unroll
            for (int e = 0; e < 8; ++e) {
                a0[e]     = xr[khalf * 8 + e];
                a0[8 + e] = xr[16 + khalf * 8 + e];
                a1[e]     = xr[32 + khalf * 8 + e];
                a1[8 + e] = xr[48 + khalf * 8 + e];
                ah[e]     = hr[khalf * 8 + e];
                ah[8 + e] = hr[16 + khalf * 8 + e];
            }
        }

        // ---- WMMA: gx (K=64, chained) and gh (K=32) -----------------------
        v8f cx = {};
        cx = __builtin_amdgcn_wmma_f32_16x16x32_bf16(false, a0, false, bw_ih0,
                                                     (short)0, cx, false, false);
        cx = __builtin_amdgcn_wmma_f32_16x16x32_bf16(false, a1, false, bw_ih1,
                                                     (short)0, cx, false, false);
        v8f ch = {};
        ch = __builtin_amdgcn_wmma_f32_16x16x32_bf16(false, ah, false, bw_hh,
                                                     (short)0, ch, false, false);

        // ---- spill C tiles (col = lane&15, elem v -> row = v + 8*khalf) ---
        #pragma unroll
        for (int v = 0; v < 8; ++v) {
            gx_s[v + 8 * khalf][wave * 16 + col] = cx[v];
            gh_s[v + 8 * khalf][wave * 16 + col] = ch[v];
        }

        // ---- store staged x_{t+1} into the other buffer -------------------
        if (more) {
            *(bf16x4*)&xb[nxt][ra][cc] = (bf16x4){(__bf16)va.x,(__bf16)va.y,(__bf16)va.z,(__bf16)va.w};
            *(bf16x4*)&xb[nxt][rb][cc] = (bf16x4){(__bf16)vb.x,(__bf16)vb.y,(__bf16)vb.z,(__bf16)vb.w};
        }
        __syncthreads();

        // ---- gate nonlinearities + h update (registers + bf16 LDS) --------
        {
            float rg = sigmoid_fast(gx_s[r0][c0] + gh_s[r0][c0] + br0);
            float zg = sigmoid_fast(gx_s[r0][c0+HID] + gh_s[r0][c0+HID] + bz0);
            float ng = tanh_fast(gx_s[r0][c0+2*HID] + bni0 + rg * (gh_s[r0][c0+2*HID] + bnh0));
            h0 = (1.f - zg) * ng + zg * h0;
            hb[r0][c0] = (__bf16)h0;
        }
        {
            float rg = sigmoid_fast(gx_s[r1][c1] + gh_s[r1][c1] + br1);
            float zg = sigmoid_fast(gx_s[r1][c1+HID] + gh_s[r1][c1+HID] + bz1);
            float ng = tanh_fast(gx_s[r1][c1+2*HID] + bni1 + rg * (gh_s[r1][c1+2*HID] + bnh1));
            h1 = (1.f - zg) * ng + zg * h1;
            hb[r1][c1] = (__bf16)h1;
        }
        if (has2) {
            float rg = sigmoid_fast(gx_s[r2][c2] + gh_s[r2][c2] + br2);
            float zg = sigmoid_fast(gx_s[r2][c2+HID] + gh_s[r2][c2+HID] + bz2);
            float ng = tanh_fast(gx_s[r2][c2+2*HID] + bni2 + rg * (gh_s[r2][c2+2*HID] + bnh2));
            h2 = (1.f - zg) * ng + zg * h2;
            hb[r2][c2] = (__bf16)h2;
        }
        __syncthreads();
    }

    // -------- head projection: out[b] = h_T . W_head + b_head --------------
    hf[r0][c0] = h0;
    hf[r1][c1] = h1;
    if (has2) hf[r2][c2] = h2;
    __syncthreads();
    if (tid < MT) {
        float s = b_head[0];
        #pragma unroll
        for (int c = 0; c < HID; ++c) s += hf[tid][c] * W_head[c];
        out[b0 + tid] = s;
    }
}

extern "C" void kernel_launch(void* const* d_in, const int* in_sizes, int n_in,
                              void* d_out, int out_size, void* d_ws, size_t ws_size,
                              hipStream_t stream) {
    (void)in_sizes; (void)n_in; (void)d_ws; (void)ws_size; (void)out_size;
    const float* x      = (const float*)d_in[0];
    const float* W_ih   = (const float*)d_in[1];
    const float* W_hh   = (const float*)d_in[2];
    const float* b_ih   = (const float*)d_in[3];
    const float* b_hh   = (const float*)d_in[4];
    const float* W_head = (const float*)d_in[5];
    const float* b_head = (const float*)d_in[6];
    float* out = (float*)d_out;

    dim3 grid(B_TOT / MT);   // 256 workgroups
    dim3 block(NTHR);        // 4 wave32s
    gru_fused_wmma<<<grid, block, 0, stream>>>(x, W_ih, W_hh, b_ih, b_hh,
                                               W_head, b_head, out);
}
